// MambaRFlayer_8512625181045
// MI455X (gfx1250) — matrix operand order repaired and checked by reading, not compiled, not run on hardware
//
#include <hip/hip_runtime.h>
#include <hip/hip_bf16.h>
#include <math.h>

// ---------------- fixed dims ----------------
#define NB   8      // batch
#define NDM  256    // d_model
#define NL   1024   // H*W
#define NR   4      // rotations
#define NDI  512    // d_inner
#define NDH  256    // d_inner/2
#define NST  16     // d_state
#define NDTR 16     // dt_rank (padded to 32 for WMMA K)

typedef __attribute__((ext_vector_type(16))) _Float16 v16h;
typedef __attribute__((ext_vector_type(8)))  _Float16 v8h;
typedef __attribute__((ext_vector_type(8)))  float    v8f;

// ---------------- helpers ----------------
__device__ __forceinline__ int rot_src(int r, int i, int j) {
    // original flat index (i0*32+j0) feeding rotated pixel (i,j) of rot90(m, r)
    switch (r & 3) {
        case 0:  return i * 32 + j;
        case 1:  return j * 32 + (31 - i);
        case 2:  return (31 - i) * 32 + (31 - j);
        default: return (31 - j) * 32 + i;
    }
}
__device__ __forceinline__ float siluf(float v)     { return v / (1.0f + expf(-v)); }
__device__ __forceinline__ float sigmf(float v)     { return 1.0f / (1.0f + expf(-v)); }
__device__ __forceinline__ float softplusf(float v) { return (v > 20.f) ? v : log1pf(expf(v)); }
__device__ __forceinline__ float geluf(float v) {
    float t = 0.7978845608028654f * (v + 0.044715f * v * v * v);
    return 0.5f * v * (1.0f + tanhf(t));
}

// A fragment piece: lane half g, element e -> k = +g*16 + e (16 consecutive halves).
__device__ __forceinline__ v16h ldA16x32(const _Float16* __restrict__ p) {
    v8h lo = *(const v8h*)p;
    v8h hi = *(const v8h*)(p + 8);
    return __builtin_shufflevector(lo, hi, 0, 1, 2, 3, 4, 5, 6, 7,
                                   8, 9, 10, 11, 12, 13, 14, 15);
}
// B fragment from N-major storage: 16 consecutive halves per lane.
__device__ __forceinline__ v16h ldB32x16(const _Float16* __restrict__ p) {
    v8h lo = *(const v8h*)p;
    v8h hi = *(const v8h*)(p + 8);
    return __builtin_shufflevector(lo, hi, 0, 1, 2, 3, 4, 5, 6, 7,
                                   8, 9, 10, 11, 12, 13, 14, 15);
}

// ---------------- generic batched WMMA GEMM: C = A * BT^T ----------------
// 16x64 tile per wave: one A fragment reused across 4 N-subtiles (4 WMMAs / k-step).
// All of a k-step's fragments are loaded before the first WMMA so the scheduler can
// stage s_wait_loadcnt and overlap global loads with matrix math.
// A:  (M,K) row-major f16, per-batch offset (z % aRepeat) * strideA
// BT: (N,K) f16 (N-major),  per-batch offset z * strideB
// C:  (M,N) f32 row-major,  per-batch offset z * strideC
// Requirements: M % 16 == 0, N % 256 == 0, K % 32 == 0, lda/ldbt % 8 == 0.
// epilogue: 0 = none, 1 = tanh-GELU
__global__ void __launch_bounds__(128) wmma_gemm_bt(
    const _Float16* __restrict__ A, const _Float16* __restrict__ BT, float* __restrict__ C,
    int Kdim, int lda, int ldbt, int ldc,
    long long strideA, int aRepeat, long long strideB, long long strideC, int epilogue)
{
    int lane = threadIdx.x & 31;
    int wave = threadIdx.x >> 5;
    int mBase = blockIdx.x << 4;
    int nBase = (blockIdx.y * 4 + wave) << 6;   // 64 columns per wave
    int z = blockIdx.z;
    const _Float16* Ab = A + (size_t)(z % aRepeat) * strideA;
    const _Float16* Bb = BT + (size_t)z * strideB;
    float* Cb = C + (size_t)z * strideC;

    const _Float16* aP = Ab + (size_t)(mBase + (lane & 15)) * lda + ((lane >> 4) << 4);
    const _Float16* bP = Bb + (size_t)(nBase + (lane & 15)) * ldbt + ((lane >> 4) << 4);
    size_t bSub = (size_t)16 * ldbt;            // stride between N-subtiles

    v8f acc0 = {}, acc1 = {}, acc2 = {}, acc3 = {};
    for (int kb = 0; kb < Kdim; kb += 32) {
        v16h a  = ldA16x32(aP + kb);
        v16h b0 = ldB32x16(bP + kb);
        v16h b1 = ldB32x16(bP + bSub + kb);
        v16h b2 = ldB32x16(bP + 2 * bSub + kb);
        v16h b3 = ldB32x16(bP + 3 * bSub + kb);
        __builtin_prefetch(aP + kb + 32, 0, 1);
        __builtin_prefetch(bP + kb + 32, 0, 1);
        acc0 = __builtin_amdgcn_wmma_f32_16x16x32_f16(false, a, false, b0, (short)0, acc0,
                                                      false, false);
        acc1 = __builtin_amdgcn_wmma_f32_16x16x32_f16(false, a, false, b1, (short)0, acc1,
                                                      false, false);
        acc2 = __builtin_amdgcn_wmma_f32_16x16x32_f16(false, a, false, b2, (short)0, acc2,
                                                      false, false);
        acc3 = __builtin_amdgcn_wmma_f32_16x16x32_f16(false, a, false, b3, (short)0, acc3,
                                                      false, false);
    }
    int half = lane >> 4;
    v8f accv[4] = {acc0, acc1, acc2, acc3};
#pragma unroll
    for (int t = 0; t < 4; ++t) {
        int n = nBase + (t << 4) + (lane & 15);
#pragma unroll
        for (int j = 0; j < 8; ++j) {
            int m = mBase + j + (half << 3);
            float v = accv[t][j];
            if (epilogue == 1) v = geluf(v);
            Cb[(size_t)m * ldc + n] = v;
        }
    }
}

// ---------------- spatial-attention 3x3 conv: 9 accumulated WMMA taps ----------------
// Wh:  sa_w1 repacked (tap, oc, ic) f16 -> per-tap A is (128,2048) row-major.
// YpT: padded y_flat, pixel-major f16 (NB, 34*34, 2048).
// Hid: (NB, 128, 1024) f32 = ReLU(conv + bias). 16x64 wave tile, A reused over 4 subtiles.
__global__ void __launch_bounds__(128) sa_conv3x3_wmma(
    const _Float16* __restrict__ Wh, const _Float16* __restrict__ YpT,
    const float* __restrict__ bias, float* __restrict__ Hid)
{
    int lane = threadIdx.x & 31;
    int wave = threadIdx.x >> 5;
    int mBase = blockIdx.x << 4;                 // 8 tiles over 128 oc
    int nBase = (blockIdx.y * 4 + wave) << 6;    // 64 pixels per wave
    int b = blockIdx.z;
    int g4 = (lane >> 4) << 4;
    const _Float16* Yb = YpT + (size_t)b * 1156 * 2048;

    size_t bOff[4];
#pragma unroll
    for (int t = 0; t < 4; ++t) {
        int nCol = nBase + (t << 4) + (lane & 15);
        int pos = ((nCol >> 5) * 34) + (nCol & 31);       // pixel -> padded i*34+j
        bOff[t] = (size_t)pos * 2048 + g4;
    }
    const _Float16* aBase = Wh + (size_t)(mBase + (lane & 15)) * 2048 + ((lane >> 4) << 4);

    v8f acc0 = {}, acc1 = {}, acc2 = {}, acc3 = {};
    for (int tap = 0; tap < 9; ++tap) {
        const _Float16* aP = aBase + (size_t)tap * 128 * 2048;
        const _Float16* Bt = Yb + (size_t)((tap / 3) * 34 + (tap % 3)) * 2048;
        for (int kb = 0; kb < 2048; kb += 32) {
            v16h a  = ldA16x32(aP + kb);
            v16h b0 = ldB32x16(Bt + bOff[0] + kb);
            v16h b1 = ldB32x16(Bt + bOff[1] + kb);
            v16h b2 = ldB32x16(Bt + bOff[2] + kb);
            v16h b3 = ldB32x16(Bt + bOff[3] + kb);
            __builtin_prefetch(aP + kb + 32, 0, 1);
            acc0 = __builtin_amdgcn_wmma_f32_16x16x32_f16(false, a, false, b0, (short)0, acc0,
                                                          false, false);
            acc1 = __builtin_amdgcn_wmma_f32_16x16x32_f16(false, a, false, b1, (short)0, acc1,
                                                          false, false);
            acc2 = __builtin_amdgcn_wmma_f32_16x16x32_f16(false, a, false, b2, (short)0, acc2,
                                                          false, false);
            acc3 = __builtin_amdgcn_wmma_f32_16x16x32_f16(false, a, false, b3, (short)0, acc3,
                                                          false, false);
        }
    }
    int half = lane >> 4;
    v8f accv[4] = {acc0, acc1, acc2, acc3};
#pragma unroll
    for (int t = 0; t < 4; ++t) {
        int nCol = nBase + (t << 4) + (lane & 15);
#pragma unroll
        for (int j = 0; j < 8; ++j) {
            int m = mBase + j + (half << 3);
            float v = accv[t][j] + bias[m];
            Hid[((size_t)b * 128 + m) * 1024 + nCol] = v > 0.f ? v : 0.f;
        }
    }
}

// ---------------- packing kernels ----------------
__global__ void pack_f16(const float* __restrict__ s, _Float16* __restrict__ d, int n) {
    int i = blockIdx.x * blockDim.x + threadIdx.x;
    if (i < n) d[i] = (_Float16)s[i];
}

// (Z, C, L) f32 -> (Z, L, C) f16   (N-major activation for WMMA B)
__global__ void pack_f16_T(const float* __restrict__ s, _Float16* __restrict__ d,
                           int Z, int C, int L) {
    int i = blockIdx.x * blockDim.x + threadIdx.x;
    if (i >= Z * C * L) return;
    int l = i % L;
    int t = i / L;
    int c = t % C;
    int z = t / C;
    d[((size_t)z * L + l) * C + c] = (_Float16)s[i];
}

// dt weights (R*256, 16) -> (R*256, 32) zero-padded f16
__global__ void pack_dtw(const float* __restrict__ w, _Float16* __restrict__ d) {
    int i = blockIdx.x * blockDim.x + threadIdx.x;
    if (i >= NR * NDH * 32) return;
    int k = i & 31, m = i >> 5;
    d[i] = (_Float16)((k < NDTR) ? w[m * NDTR + k] : 0.f);
}

// dts rows (k<16) of x_dbl (32 blocks of (48, NL)) -> transposed padded (32, NL, 32) f16
__global__ void pack_dts_T(const float* __restrict__ xdbl, _Float16* __restrict__ d) {
    int i = blockIdx.x * blockDim.x + threadIdx.x;
    if (i >= NB * NR * NL * 32) return;
    int k = i & 31;
    int t = i >> 5;
    int l = t % NL;
    int z = t / NL;
    d[i] = (_Float16)((k < NDTR) ? xdbl[((size_t)z * 48 + k) * NL + l] : 0.f);
}

// sa_w1 (oc, ic, 3, 3) -> (tap, oc, ic) f16
__global__ void pack_saw1(const float* __restrict__ w, _Float16* __restrict__ d) {
    int i = blockIdx.x * blockDim.x + threadIdx.x;
    if (i >= 9 * 128 * 2048) return;
    int ic = i & 2047;
    int t = i >> 11;
    int oc = t % 128;
    int tap = t / 128;
    d[i] = (_Float16)w[((size_t)oc * 2048 + ic) * 9 + tap];
}

// zero-padded, pixel-major f16 image (NB, 34*34, 2048) from ybuf (NB, 2048, 32, 32)
__global__ void build_ypad_T(const float* __restrict__ Y, _Float16* __restrict__ Yp, int total) {
    int idx = blockIdx.x * blockDim.x + threadIdx.x;
    if (idx >= total) return;
    int c = idx & 2047;
    int t = idx >> 11;
    int pos = t % 1156;
    int b = t / 1156;
    int i = pos / 34 - 1, j = pos % 34 - 1;
    float v = 0.f;
    if (i >= 0 && i < 32 && j >= 0 && j < 32)
        v = Y[(((size_t)b * 2048 + c) * NL) + i * 32 + j];
    Yp[idx] = (_Float16)v;
}

// ---------------- elementwise / sequential kernels ----------------
// rotation-aware depthwise conv1d (K=4, 'same') + SiLU on both halves of xz.
// x branch -> xseq f32 (seq-major) + xseqT f16 (N-major); z branch -> un-rotated into ybuf.
__global__ void dwconv_rot(const float* __restrict__ xz,
                           const float* __restrict__ wx, const float* __restrict__ bx,
                           const float* __restrict__ wz, const float* __restrict__ bz,
                           float* __restrict__ xseq, _Float16* __restrict__ xseqT,
                           float* __restrict__ ybuf)
{
    int tid = blockIdx.x * blockDim.x + threadIdx.x;  // over NB*NR*NDH
    if (tid >= NB * NR * NDH) return;
    int ch = tid % NDH;
    int t = tid / NDH;
    int r = t % NR, b = t / NR;
    int cg = r * NDH + ch;
    const float* xc = xz + ((size_t)b * NDI + ch) * NL;
    const float* zc = xz + ((size_t)b * NDI + NDH + ch) * NL;
    float wxa[4], wza[4];
#pragma unroll
    for (int k = 0; k < 4; ++k) { wxa[k] = wx[cg * 4 + k]; wza[k] = wz[cg * 4 + k]; }
    float bxv = bx[cg], bzv = bz[cg];
    size_t obase = ((size_t)(b * NR + r) * NDH + ch) * NL;
    size_t tbase = (size_t)(b * NR + r) * NL * NDH + ch;
    size_t zbase = ((size_t)(b * NR + r) * NDI + NDH + ch) * NL;
    for (int l = 0; l < NL; ++l) {
        float sx = bxv, sz = bzv;
#pragma unroll
        for (int tp = 0; tp < 4; ++tp) {
            int p = l + tp - 1;
            if (p >= 0 && p < NL) {
                int s = rot_src(r, p >> 5, p & 31);
                sx += wxa[tp] * xc[s];
                sz += wza[tp] * zc[s];
            }
        }
        sx = siluf(sx);
        sz = siluf(sz);
        xseq[obase + l] = sx;
        xseqT[tbase + (size_t)l * NDH] = (_Float16)sx;
        ybuf[zbase + rot_src(r, l >> 5, l & 31)] = sz;
    }
}

// selective scan: one thread per (b, r, d); writes y (+D*u) un-rotated into ybuf[ch 0..DH)
__global__ void selective_scan(const float* __restrict__ dpre, const float* __restrict__ dtb,
                               const float* __restrict__ Alogs, const float* __restrict__ Dsv,
                               const float* __restrict__ xseq, const float* __restrict__ xdbl,
                               float* __restrict__ ybuf)
{
    int tid = blockIdx.x * blockDim.x + threadIdx.x;
    if (tid >= NB * NR * NDH) return;
    int d = tid % NDH;
    int t = tid / NDH;
    int r = t % NR, b = t / NR;
    int rd = r * NDH + d;
    float a[NST], h[NST];
#pragma unroll
    for (int n = 0; n < NST; ++n) { a[n] = -expf(Alogs[rd * NST + n]); h[n] = 0.f; }
    float Dv = Dsv[rd];
    float bias = dtb[rd];
    const float* dp = dpre + ((size_t)(b * NR + r) * NDH + d) * NL;
    const float* u  = xseq + ((size_t)(b * NR + r) * NDH + d) * NL;
    const float* Bm = xdbl + ((size_t)(b * NR + r) * 48 + NDTR) * NL;
    const float* Cm = xdbl + ((size_t)(b * NR + r) * 48 + NDTR + NST) * NL;
    size_t ob = ((size_t)(b * NR + r) * NDI + d) * NL;
    for (int l = 0; l < NL; ++l) {
        float dt = softplusf(dp[l] + bias);
        float uv = u[l];
        float du = dt * uv;
        float y = 0.f;
#pragma unroll
        for (int n = 0; n < NST; ++n) {
            h[n] = expf(dt * a[n]) * h[n] + du * Bm[(size_t)n * NL + l];
            y += h[n] * Cm[(size_t)n * NL + l];
        }
        y += Dv * uv;
        ybuf[ob + rot_src(r, l >> 5, l & 31)] = y;
    }
}

// sa conv2 (128->4) + softmax over rotations, per pixel
__global__ void spatial_attn(const float* __restrict__ Hid, const float* __restrict__ w2,
                             const float* __restrict__ b2, float* __restrict__ sw)
{
    int tid = blockIdx.x * blockDim.x + threadIdx.x;  // NB*NL
    if (tid >= NB * NL) return;
    int l = tid & (NL - 1);
    int b = tid >> 10;
    float lg[NR];
    float mx = -1e30f;
#pragma unroll
    for (int r = 0; r < NR; ++r) {
        float s = b2[r];
        for (int h = 0; h < 128; ++h) s += w2[r * 128 + h] * Hid[((size_t)b * 128 + h) * NL + l];
        lg[r] = s;
        mx = fmaxf(mx, s);
    }
    float den = 0.f;
#pragma unroll
    for (int r = 0; r < NR; ++r) { lg[r] = expf(lg[r] - mx); den += lg[r]; }
#pragma unroll
    for (int r = 0; r < NR; ++r) sw[((size_t)b * NR + r) * NL + l] = lg[r] / den;
}

// channel mean over H*W of y_flat
__global__ void chan_avg(const float* __restrict__ Y, float* __restrict__ cavg) {
    int tid = blockIdx.x * blockDim.x + threadIdx.x;  // NB*2048
    if (tid >= NB * 2048) return;
    const float* p = Y + (size_t)tid * NL;
    float s = 0.f;
    for (int l = 0; l < NL; ++l) s += p[l];
    cavg[tid] = s * (1.0f / NL);
}

// channel attention MLP + softmax -> cw (NB, NR)
__global__ void chan_attn(const float* __restrict__ cavg, const float* __restrict__ w1,
                          const float* __restrict__ w2, float* __restrict__ cw)
{
    __shared__ float hid[128];
    int b = blockIdx.x;
    int h = threadIdx.x;
    if (h < 128) {
        float s = 0.f;
        const float* cv = cavg + (size_t)b * 2048;
        for (int c = 0; c < 2048; ++c) s += w1[(size_t)h * 2048 + c] * cv[c];
        hid[h] = s > 0.f ? s : 0.f;
    }
    __syncthreads();
    if (h == 0) {
        float lg[NR];
        float mx = -1e30f;
#pragma unroll
        for (int r = 0; r < NR; ++r) {
            float s = 0.f;
            for (int k = 0; k < 128; ++k) s += w2[r * 128 + k] * hid[k];
            lg[r] = s;
            mx = fmaxf(mx, s);
        }
        float den = 0.f;
#pragma unroll
        for (int r = 0; r < NR; ++r) { lg[r] = expf(lg[r] - mx); den += lg[r]; }
#pragma unroll
        for (int r = 0; r < NR; ++r) cw[b * NR + r] = lg[r] / den;
    }
}

// spatial/channel weighted sums (f32, row-major) + N-major f16 concat for gate GEMM
__global__ void mix_branches(const float* __restrict__ ybuf, const float* __restrict__ sw,
                             const float* __restrict__ cw, float* __restrict__ spat,
                             float* __restrict__ chan, _Float16* __restrict__ catT)
{
    int tid = blockIdx.x * blockDim.x + threadIdx.x;  // NB*NDI*NL
    if (tid >= NB * NDI * NL) return;
    int l = tid & (NL - 1);
    int t = tid >> 10;
    int c = t % NDI;
    int b = t / NDI;
    float s = 0.f, ch = 0.f;
#pragma unroll
    for (int r = 0; r < NR; ++r) {
        float v = ybuf[((size_t)(b * NR + r) * NDI + c) * NL + l];
        s += v * sw[((size_t)b * NR + r) * NL + l];
        ch += v * cw[b * NR + r];
    }
    size_t o = ((size_t)b * NDI + c) * NL + l;
    spat[o] = s;
    chan[o] = ch;
    size_t tb = ((size_t)b * NL + l) * (2 * NDI) + c;
    catT[tb] = (_Float16)s;
    catT[tb + NDI] = (_Float16)ch;
}

// fused = sigmoid(gpre + gate_b) * spatial + (1-g) * channel -> N-major f16 for out_proj
__global__ void gate_fuse(const float* __restrict__ gpre, const float* __restrict__ gb,
                          const float* __restrict__ spat, const float* __restrict__ chan,
                          _Float16* __restrict__ fusedT)
{
    int tid = blockIdx.x * blockDim.x + threadIdx.x;  // NB*NDI*NL
    if (tid >= NB * NDI * NL) return;
    int l = tid & (NL - 1);
    int t = tid >> 10;
    int c = t % NDI;
    int b = t / NDI;
    float g = sigmf(gpre[tid] + gb[c]);
    fusedT[((size_t)b * NL + l) * NDI + c] = (_Float16)(g * spat[tid] + (1.f - g) * chan[tid]);
}

// ---------------- host orchestration ----------------
extern "C" void kernel_launch(void* const* d_in, const int* in_sizes, int n_in,
                              void* d_out, int out_size, void* d_ws, size_t ws_size,
                              hipStream_t stream) {
    const float* x               = (const float*)d_in[0];
    const float* in_proj_w       = (const float*)d_in[1];
    const float* out_proj_w      = (const float*)d_in[2];
    const float* conv_x_w        = (const float*)d_in[3];
    const float* conv_x_b        = (const float*)d_in[4];
    const float* conv_z_w        = (const float*)d_in[5];
    const float* conv_z_b        = (const float*)d_in[6];
    const float* x_proj_weight   = (const float*)d_in[7];
    const float* dt_projs_weight = (const float*)d_in[8];
    const float* dt_projs_bias   = (const float*)d_in[9];
    const float* A_logs          = (const float*)d_in[10];
    const float* Dsv             = (const float*)d_in[11];
    const float* sa_w1           = (const float*)d_in[12];
    const float* sa_b1           = (const float*)d_in[13];
    const float* sa_w2           = (const float*)d_in[14];
    const float* sa_b2           = (const float*)d_in[15];
    const float* ca_w1           = (const float*)d_in[16];
    const float* ca_w2           = (const float*)d_in[17];
    const float* gate_w          = (const float*)d_in[18];
    const float* gate_b          = (const float*)d_in[19];
    float* out = (float*)d_out;

    char* p = (char*)d_ws;
    auto alloc = [&](size_t bytes) -> void* {
        void* r = (void*)p;
        p += (bytes + 255) & ~(size_t)255;
        return r;
    };
    _Float16* w_in_h   = (_Float16*)alloc((size_t)NDI * NDM * 2);
    _Float16* w_xp_h   = (_Float16*)alloc((size_t)NR * 48 * NDH * 2);
    _Float16* w_dt_h   = (_Float16*)alloc((size_t)NR * NDH * 32 * 2);       // K padded to 32
    _Float16* w_sa1_h  = (_Float16*)alloc((size_t)9 * 128 * 2048 * 2);      // (tap, oc, ic)
    _Float16* w_gate_h = (_Float16*)alloc((size_t)NDI * 2 * NDI * 2);
    _Float16* w_out_h  = (_Float16*)alloc((size_t)NDM * NDI * 2);
    _Float16* x_T      = (_Float16*)alloc((size_t)NB * NL * NDM * 2);       // (b, l, c)
    float*    xz       = (float*)alloc((size_t)NB * NDI * NL * 4);
    float*    xseq     = (float*)alloc((size_t)NB * NR * NDH * NL * 4);
    _Float16* xseq_T   = (_Float16*)alloc((size_t)NB * NR * NL * NDH * 2);  // (b,r,l,ch)
    float*    xdbl     = (float*)alloc((size_t)NB * NR * 48 * NL * 4);
    _Float16* dts_T    = (_Float16*)alloc((size_t)NB * NR * NL * 32 * 2);   // (z, l, 32)
    float*    dpre     = (float*)alloc((size_t)NB * NR * NDH * NL * 4);
    float*    ybuf     = (float*)alloc((size_t)NB * NR * NDI * NL * 4);
    _Float16* ypad_T   = (_Float16*)alloc((size_t)NB * 1156 * 2048 * 2);    // (b, pos, ic)
    float*    hid      = (float*)alloc((size_t)NB * 128 * NL * 4);
    float*    swb      = (float*)alloc((size_t)NB * NR * NL * 4);
    float*    cavg     = (float*)alloc((size_t)NB * 2048 * 4);
    float*    cwb      = (float*)alloc((size_t)NB * NR * 4);
    float*    spat     = (float*)alloc((size_t)NB * NDI * NL * 4);
    float*    chan     = (float*)alloc((size_t)NB * NDI * NL * 4);
    _Float16* cat_T    = (_Float16*)alloc((size_t)NB * NL * 2 * NDI * 2);   // (b, l, 1024)
    float*    gpre     = (float*)alloc((size_t)NB * NDI * NL * 4);
    _Float16* fused_T  = (_Float16*)alloc((size_t)NB * NL * NDI * 2);       // (b, l, 512)
    (void)ws_size; (void)n_in; (void)in_sizes; (void)out_size;

    auto pk = [&](const float* s, _Float16* d, int n) {
        pack_f16<<<(n + 255) / 256, 256, 0, stream>>>(s, d, n);
    };
    pk(in_proj_w, w_in_h, NDI * NDM);
    pk(x_proj_weight, w_xp_h, NR * 48 * NDH);
    pk(gate_w, w_gate_h, NDI * 2 * NDI);
    pk(out_proj_w, w_out_h, NDM * NDI);
    pack_dtw<<<(NR * NDH * 32 + 255) / 256, 256, 0, stream>>>(dt_projs_weight, w_dt_h);
    pack_saw1<<<(9 * 128 * 2048 + 255) / 256, 256, 0, stream>>>(sa_w1, w_sa1_h);
    pack_f16_T<<<(NB * NDM * NL + 255) / 256, 256, 0, stream>>>(x, x_T, NB, NDM, NL);

    // in_proj: per-batch (512x256) x (256x1024)
    wmma_gemm_bt<<<dim3(NDI / 16, 4, NB), 128, 0, stream>>>(
        w_in_h, x_T, xz, NDM, NDM, NDM, NL,
        0LL, 1, (long long)NL * NDM, (long long)NDI * NL, 0);

    // rotation-aware depthwise conv + SiLU; z-branch scattered into ybuf
    dwconv_rot<<<(NB * NR * NDH + 255) / 256, 256, 0, stream>>>(
        xz, conv_x_w, conv_x_b, conv_z_w, conv_z_b, xseq, xseq_T, ybuf);

    // x_proj: per-(b,r) (48x256) x (256x1024)
    wmma_gemm_bt<<<dim3(48 / 16, 4, NB * NR), 128, 0, stream>>>(
        w_xp_h, xseq_T, xdbl, NDH, NDH, NDH, NL,
        (long long)48 * NDH, NR, (long long)NL * NDH, (long long)48 * NL, 0);

    // dts rows -> N-major f16, K zero-padded 16 -> 32
    pack_dts_T<<<(NB * NR * NL * 32 + 255) / 256, 256, 0, stream>>>(xdbl, dts_T);

    // dt_proj: per-(b,r) (256x32) x (32x1024)   (upper 16 of K are zeros)
    wmma_gemm_bt<<<dim3(NDH / 16, 4, NB * NR), 128, 0, stream>>>(
        w_dt_h, dts_T, dpre, 32, 32, 32, NL,
        (long long)NDH * 32, NR, (long long)NL * 32, (long long)NDH * NL, 0);

    // selective scan + D*u, un-rotated scatter into ybuf channels [0, DH)
    selective_scan<<<(NB * NR * NDH + 255) / 256, 256, 0, stream>>>(
        dpre, dt_projs_bias, A_logs, Dsv, xseq, xdbl, ybuf);

    // padded pixel-major f16 image for the 3x3 conv
    {
        int total = NB * 1156 * 2048;
        build_ypad_T<<<(total + 255) / 256, 256, 0, stream>>>(ybuf, ypad_T, total);
    }

    // spatial attention conv1 (3x3, 2048->128) via 9 WMMA taps, bias+ReLU
    sa_conv3x3_wmma<<<dim3(8, 4, NB), 128, 0, stream>>>(w_sa1_h, ypad_T, sa_b1, hid);

    // conv2 (128->4) + softmax over rotations
    spatial_attn<<<(NB * NL + 255) / 256, 256, 0, stream>>>(hid, sa_w2, sa_b2, swb);

    // channel attention
    chan_avg<<<(NB * 2048 + 255) / 256, 256, 0, stream>>>(ybuf, cavg);
    chan_attn<<<NB, 128, 0, stream>>>(cavg, ca_w1, ca_w2, cwb);

    // spatial & channel weighted sums + N-major concat (f16) for gate
    mix_branches<<<(NB * NDI * NL + 255) / 256, 256, 0, stream>>>(
        ybuf, swb, cwb, spat, chan, cat_T);

    // gate 1x1 conv: per-batch (512x1024) x (1024x1024)
    wmma_gemm_bt<<<dim3(NDI / 16, 4, NB), 128, 0, stream>>>(
        w_gate_h, cat_T, gpre, 2 * NDI, 2 * NDI, 2 * NDI, NL,
        0LL, 1, (long long)NL * 2 * NDI, (long long)NDI * NL, 0);

    // sigmoid gate + blend -> N-major f16
    gate_fuse<<<(NB * NDI * NL + 255) / 256, 256, 0, stream>>>(
        gpre, gate_b, spat, chan, fused_T);

    // out_proj: per-batch (256x512) x (512x1024), GELU epilogue, straight to d_out
    wmma_gemm_bt<<<dim3(NDM / 16, 4, NB), 128, 0, stream>>>(
        w_out_h, fused_T, out, NDI, NDI, NDI, NL,
        0LL, 1, (long long)NL * NDI, (long long)NDM * NL, 1);
}
